// RCNModel_53188874994147
// MI455X (gfx1250) — matrix operand order repaired and checked by reading, not compiled, not run on hardware
//
#include <hip/hip_runtime.h>
#include <hip/hip_bf16.h>

// ---------------------------------------------------------------------------
// GATv2 GNN forward for MI455X (gfx1250, wave32).
//  * Dense transforms: v_wmma_f32_16x16x32_bf16, 16x64 output strip per wave,
//    bf16-pre-converted A (two b128 loads/lane/kstep) and B pre-packed into
//    exact fragment order (contiguous 32B/lane) -> no divergence, no per-load
//    waits in the hot loop.
//  * Edge softmax/aggregation: f32 global atomics; bandwidth bound
//    (~6-8 GB -> ~300us at 23.3 TB/s). GEMMs are noise on the matrix cores.
// ---------------------------------------------------------------------------

typedef __attribute__((ext_vector_type(16))) __bf16 v16bf;
typedef __attribute__((ext_vector_type(8)))  __bf16 v8bf;
typedef __attribute__((ext_vector_type(8)))  float  v8f;

#define WAVE 32

#define WMMA_BF16(A, B, C) \
    __builtin_amdgcn_wmma_f32_16x16x32_bf16(false, (A), false, (B), (short)0, (C), false, false)

// --- pre-passes -------------------------------------------------------------

// X[n,K] f32 -> out[n,Kpad] bf16 (zero pad K..Kpad)
__global__ void k_cvt_pad_bf16(const float* __restrict__ X, __bf16* __restrict__ out,
                               int n, int K, int Kpad) {
    long i = blockIdx.x * (long)blockDim.x + threadIdx.x;
    if (i >= (long)n * Kpad) return;
    int k = (int)(i % Kpad);
    long r = i / Kpad;
    out[i] = (k < K) ? (__bf16)X[r * K + k] : (__bf16)0.0f;
}

// W[K,M] f32 -> B-fragment-packed bf16.
// fragment elem (ct, ks, lane, e): k = ks*32 + (e&7) + (e>=8?16:0) + (lane>=16?8:0), col = ct*16 + (lane&15)
// stored contiguously at (((ct*ksteps + ks)*32) + lane)*16 + e  -> 32B/lane loads.
__global__ void k_pack_w_bf16(const float* __restrict__ W, __bf16* __restrict__ out,
                              int K, int Kpad, int M) {
    int ksteps = Kpad >> 5;
    long total = (long)(M >> 4) * ksteps * 32 * 16;
    long i = blockIdx.x * (long)blockDim.x + threadIdx.x;
    if (i >= total) return;
    int e    = (int)(i & 15);
    int lane = (int)((i >> 4) & 31);
    long rest = i >> 9;                       // ct*ksteps + ks
    int ct = (int)(rest / ksteps);
    int ks = (int)(rest - (long)ct * ksteps);
    int k   = (ks << 5) + (e & 7) + ((e & 8) ? 16 : 0) + ((lane >= 16) ? 8 : 0);
    int col = (ct << 4) + (lane & 15);
    out[i] = (k < K) ? (__bf16)W[(long)k * M + col] : (__bf16)0.0f;
}

// --- WMMA GEMM: Y[n,M] = act(A(bf16)[n,Kpad] @ W + bias) --------------------
// requires n%16==0, M%64==0, Kpad%32==0 (guaranteed by caller/padding).
__global__ __launch_bounds__(256)
void k_gemm_bf16(const __bf16* __restrict__ A, const __bf16* __restrict__ Bp,
                 const float* __restrict__ bias, float* __restrict__ Y,
                 int n, int Kpad, int M, int act /*0 none, 1 relu*/) {
    int wave = (int)((blockIdx.x * (unsigned)blockDim.x + threadIdx.x) >> 5);
    int lane = threadIdx.x & 31;
    int colStrips = M >> 6;                   // 4 x 16-col tiles per wave
    int strips = (n >> 4) * colStrips;
    if (wave >= strips) return;               // wave-uniform
    int rt = wave / colStrips, cs = wave - rt * colStrips;
    int row0 = rt << 4;
    int ct0  = cs << 2;
    int ksteps = Kpad >> 5;
    const __bf16* Arow = A + (long)(row0 + (lane & 15)) * Kpad + ((lane >= 16) ? 8 : 0);
    const __bf16* Bl   = Bp + ((((long)ct0 * ksteps) << 5) + lane) * 16;
    const long bstride = (long)ksteps << 9;   // elems per 16-col tile
    v8f acc0 = {}, acc1 = {}, acc2 = {}, acc3 = {};
    for (int ks = 0; ks < ksteps; ++ks) {
        // A fragment: ISA 16-bit 16x32 layout -> two contiguous 16B loads
        v8bf alo = *(const v8bf*)(Arow + (ks << 5));
        v8bf ahi = *(const v8bf*)(Arow + (ks << 5) + 16);
        v16bf a = __builtin_shufflevector(alo, ahi,
                   0, 1, 2, 3, 4, 5, 6, 7, 8, 9, 10, 11, 12, 13, 14, 15);
        const __bf16* bp = Bl + ((long)ks << 9);
        v16bf b0 = *(const v16bf*)(bp);
        v16bf b1 = *(const v16bf*)(bp + bstride);
        v16bf b2 = *(const v16bf*)(bp + 2 * bstride);
        v16bf b3 = *(const v16bf*)(bp + 3 * bstride);
        acc0 = WMMA_BF16(a, b0, acc0);
        acc1 = WMMA_BF16(a, b1, acc1);
        acc2 = WMMA_BF16(a, b2, acc2);
        acc3 = WMMA_BF16(a, b3, acc3);
    }
    int rbase = row0 + ((lane >= 16) ? 8 : 0);
#define STORE_ACC(T, ACC)                                            \
    {                                                                \
        int col = ((ct0 + (T)) << 4) + (lane & 15);                  \
        float bv = bias[col];                                        \
        _Pragma("unroll")                                            \
        for (int r = 0; r < 8; ++r) {                                \
            float v = (ACC)[r] + bv;                                 \
            if (act == 1) v = fmaxf(v, 0.0f);                        \
            Y[(long)(rbase + r) * M + col] = v;                      \
        }                                                            \
    }
    STORE_ACC(0, acc0) STORE_ACC(1, acc1) STORE_ACC(2, acc2) STORE_ACC(3, acc3)
#undef STORE_ACC
}

// --- utility fills ----------------------------------------------------------

__global__ void k_fill_f32(float* p, float v, long n) {
    long i = blockIdx.x * (long)blockDim.x + threadIdx.x;
    if (i < n) p[i] = v;
}
__global__ void k_fill_i32(int* p, int v, long n) {
    long i = blockIdx.x * (long)blockDim.x + threadIdx.x;
    if (i < n) p[i] = v;
}

// --- edge kernels -----------------------------------------------------------

// alpha[e,h] = sum_c leakyrelu(xl[src,h,c]+xr[dst,h,c]+ea@We, 0.2)*att[h,c]
__global__ __launch_bounds__(256)
void k_edge_alpha(const float* __restrict__ xl, const float* __restrict__ xr,
                  const float* __restrict__ eattr, const float* __restrict__ We,
                  const float* __restrict__ att,
                  const int* __restrict__ src, const int* __restrict__ dst,
                  float* __restrict__ alpha, int* __restrict__ amax,
                  int E, int H, int C) {
    long wave = (blockIdx.x * (long)blockDim.x + threadIdx.x) >> 5;
    int lane = threadIdx.x & 31;
    if (wave >= (long)E * H) return;
    int e = (int)(wave / H), h = (int)(wave - (long)(wave / H) * H);
    int s = src[e], d = dst[e];
    int HC = H * C;
    float ea0 = eattr[2 * e], ea1 = eattr[2 * e + 1];
    float sum = 0.0f;
    for (int c = lane; c < C; c += WAVE) {
        int j = h * C + c;
        float m = xl[(long)s * HC + j] + xr[(long)d * HC + j]
                + ea0 * We[j] + ea1 * We[HC + j];
        m = (m > 0.0f) ? m : 0.2f * m;
        sum += m * att[j];
    }
    for (int off = 16; off; off >>= 1) sum += __shfl_down(sum, off, WAVE);
    if (lane == 0) {
        alpha[(long)e * H + h] = sum;
        int bits = __float_as_int(sum);
        int key = (bits >= 0) ? bits : (bits ^ 0x7fffffff);   // order-preserving
        atomicMax(&amax[(long)d * H + h], key);
    }
}

__global__ void k_edge_expsum(float* __restrict__ alpha,
                              const int* __restrict__ amax,
                              const int* __restrict__ dst,
                              float* __restrict__ denom, int E, int H) {
    long i = blockIdx.x * (long)blockDim.x + threadIdx.x;
    if (i >= (long)E * H) return;
    int e = (int)(i / H), h = (int)(i - (long)e * H);
    int d = dst[e];
    int k = amax[(long)d * H + h];
    float am = (k >= 0) ? __int_as_float(k) : __int_as_float(k ^ 0x7fffffff);
    float ex = __expf(alpha[i] - am);
    alpha[i] = ex;
    atomicAdd(&denom[(long)d * H + h], ex);
}

// out[dst,h,c] += xl[src,h,c] * ex[e,h]/(denom[dst,h]+1e-16)
__global__ void k_edge_aggregate(const float* __restrict__ xl,
                                 const float* __restrict__ ex,
                                 const float* __restrict__ denom,
                                 const int* __restrict__ src,
                                 const int* __restrict__ dst,
                                 float* __restrict__ out, int E, int H, int C) {
    long HC = (long)H * C;
    long i = blockIdx.x * (long)blockDim.x + threadIdx.x;
    if (i >= (long)E * HC) return;
    int e = (int)(i / HC);
    int j = (int)(i - (long)e * HC);
    int h = j / C;
    int s = src[e], d = dst[e];
    float a = ex[(long)e * H + h] / (denom[(long)d * H + h] + 1e-16f);
    atomicAdd(&out[(long)d * HC + j], xl[(long)s * HC + j] * a);
}

// --- BN / pooling / heads ---------------------------------------------------

__global__ __launch_bounds__(256)
void k_colstats(const float* __restrict__ X, const float* __restrict__ bias,
                float* __restrict__ mu, float* __restrict__ var, int n, int M) {
    int col = blockIdx.x;
    float s = 0.0f, s2 = 0.0f;
    for (int r = threadIdx.x; r < n; r += blockDim.x) {
        float v = X[(long)r * M + col] + bias[col];
        s += v; s2 += v * v;
    }
    __shared__ float sh[256], sh2[256];
    sh[threadIdx.x] = s; sh2[threadIdx.x] = s2; __syncthreads();
    for (int o = 128; o; o >>= 1) {
        if (threadIdx.x < o) {
            sh[threadIdx.x]  += sh[threadIdx.x + o];
            sh2[threadIdx.x] += sh2[threadIdx.x + o];
        }
        __syncthreads();
    }
    if (threadIdx.x == 0) {
        float m = sh[0] / n;
        mu[col] = m;
        var[col] = sh2[0] / n - m * m;
    }
}

__global__ void k_bn_elu(const float* __restrict__ X, const float* __restrict__ bias,
                         const float* __restrict__ mu, const float* __restrict__ var,
                         const float* __restrict__ g, const float* __restrict__ b,
                         float* __restrict__ Y, long n, int M) {
    long i = blockIdx.x * (long)blockDim.x + threadIdx.x;
    if (i >= n * M) return;
    int col = (int)(i % M);
    float v = X[i] + bias[col];
    v = (v - mu[col]) * rsqrtf(var[col] + 1e-5f) * g[col] + b[col];
    Y[i] = (v > 0.0f) ? v : (__expf(v) - 1.0f);   // ELU
}

__global__ void k_pool(const float* __restrict__ feat, const int* __restrict__ batch,
                       float* __restrict__ gemb, float* __restrict__ cnt,
                       int n, int M) {
    long i = blockIdx.x * (long)blockDim.x + threadIdx.x;
    if (i >= (long)n * M) return;
    int r = (int)(i / M), c = (int)(i - (long)r * M);
    int gidx = batch[r];
    atomicAdd(&gemb[(long)gidx * M + c], feat[i]);
    if (c == 0) atomicAdd(&cnt[gidx], 1.0f);
}
__global__ void k_pool_div(float* gemb, const float* cnt, int G, int M) {
    long i = blockIdx.x * (long)blockDim.x + threadIdx.x;
    if (i >= (long)G * M) return;
    gemb[i] /= fmaxf(cnt[i / M], 1.0f);
}

// out[row] = (mode ? tanh : id)(X[row,:K] . w + b[0]) ; one wave per row
__global__ __launch_bounds__(256)
void k_dot(const float* __restrict__ X, int K, const float* __restrict__ w,
           const float* __restrict__ b, float* __restrict__ out, int n, int mode) {
    int wave = (int)((blockIdx.x * (unsigned)blockDim.x + threadIdx.x) >> 5);
    int lane = threadIdx.x & 31;
    if (wave >= n) return;
    float s = 0.0f;
    for (int k = lane; k < K; k += WAVE) s += X[(long)wave * K + k] * w[k];
    for (int off = 16; off; off >>= 1) s += __shfl_down(s, off, WAVE);
    if (lane == 0) {
        s += b[0];
        out[wave] = mode ? tanhf(s) : s;
    }
}

// ---------------------------------------------------------------------------

static inline dim3 g1(long n, int t = 256) { return dim3((unsigned)((n + t - 1) / t)); }

extern "C" void kernel_launch(void* const* d_in, const int* in_sizes, int n_in,
                              void* d_out, int out_size, void* d_ws, size_t ws_size,
                              hipStream_t stream) {
    // setup_inputs dict insertion order; params flattened DFS insertion order:
    // 0:x 1:edge_attr 2..38:params 39:edge_index 40:batch_idx
    const float* x     = (const float*)d_in[0];
    const float* eattr = (const float*)d_in[1];
    int pi = 2;
    auto F = [&](void) { return (const float*)d_in[pi++]; };
    const float *c1Wl=F(), *c1bl=F(), *c1Wr=F(), *c1br=F(), *c1We=F(), *c1att=F(), *c1b=F();
    const float *c2Wl=F(), *c2bl=F(), *c2Wr=F(), *c2br=F(), *c2We=F(), *c2att=F(), *c2b=F();
    const float *c3Wl=F(), *c3bl=F(), *c3Wr=F(), *c3br=F(), *c3We=F(), *c3att=F(), *c3b=F();
    const float *g1p=F(), *b1p=F(), *g2p=F(), *b2p=F(), *g3p=F(), *b3p=F();
    const float *Wv1=F(), *bv1=F(), *Wv2=F(), *bv2=F();
    const float *Wp=F(),  *bp=F();
    const float *Wt=F(),  *bt=F(), *Ws=F(), *bs=F();
    const int* ei    = (const int*)d_in[pi++];
    const int* batch = (const int*)d_in[pi++];

    const int N = in_sizes[0] / 15;
    const int E = in_sizes[1] / 2;
    const int G = 64;
    const int* srcI = ei;
    const int* dstI = ei + E;

    // workspace carve-up (256B-aligned chunks)
    char* base = (char*)d_ws;
    size_t off = 0;
    auto carve = [&](size_t bytes) -> void* {
        void* p = base + off;
        off += (bytes + 255) & ~(size_t)255;
        return p;
    };
    float*  bufH  = (float*)carve((size_t)N * 256 * 4);
    float*  bufXL = (float*)carve((size_t)N * 256 * 4);
    float*  bufXR = (float*)carve((size_t)N * 256 * 4);   // reused as agg out
    float*  alpha = (float*)carve((size_t)E * 4 * 4);
    int*    amax  = (int*)  carve((size_t)N * 4 * 4);
    float*  denom = (float*)carve((size_t)N * 4 * 4);
    float*  muB   = (float*)carve(256 * 4);
    float*  varB  = (float*)carve(256 * 4);
    float*  gemb  = (float*)carve(((size_t)G * 256 + G) * 4);  // gemb + cnt contiguous
    float*  cnt   = gemb + (size_t)G * 256;
    float*  hid   = (float*)carve((size_t)G * 128 * 4);
    __bf16* Abf   = (__bf16*)carve((size_t)N * 256 * 2);
    __bf16* Bpk   = (__bf16*)carve((size_t)256 * 4096 * 2);

    auto pack_gemm = [&](const __bf16* Abf_, int n, int K, int Kpad,
                         const float* W, const float* bias, float* Y, int M, int act) {
        long pk = (long)(M >> 4) * (Kpad >> 5) * 32 * 16;
        k_pack_w_bf16<<<g1(pk), 256, 0, stream>>>(W, Bpk, K, Kpad, M);
        long strips = (long)(n >> 4) * (M >> 6);
        k_gemm_bf16<<<g1((strips + 7) / 8, 1), 256, 0, stream>>>(Abf_, Bpk, bias, Y, n, Kpad, M, act);
    };

    auto gat_layer = [&](const float* Xin, int Cin, int Cpad,
                         const float* Wl, const float* bl,
                         const float* Wr, const float* br,
                         const float* We, const float* att, const float* bias,
                         const float* gn, const float* bn,
                         int H, int C, float* Hout) {
        const int HC = H * C;
        k_cvt_pad_bf16<<<g1((long)N * Cpad), 256, 0, stream>>>(Xin, Abf, N, Cin, Cpad);
        pack_gemm(Abf, N, Cin, Cpad, Wl, bl, bufXL, HC, 0);
        pack_gemm(Abf, N, Cin, Cpad, Wr, br, bufXR, HC, 0);
        k_fill_i32<<<g1((long)N * H), 256, 0, stream>>>(amax, (int)0x80000000, (long)N * H);
        k_fill_f32<<<g1((long)N * H), 256, 0, stream>>>(denom, 0.0f, (long)N * H);
        k_edge_alpha<<<g1(((long)E * H + 7) / 8, 1), 256, 0, stream>>>(
            bufXL, bufXR, eattr, We, att, srcI, dstI, alpha, amax, E, H, C);
        k_edge_expsum<<<g1((long)E * H), 256, 0, stream>>>(alpha, amax, dstI, denom, E, H);
        k_fill_f32<<<g1((long)N * HC), 256, 0, stream>>>(bufXR, 0.0f, (long)N * HC);
        k_edge_aggregate<<<g1((long)E * HC), 256, 0, stream>>>(
            bufXL, alpha, denom, srcI, dstI, bufXR, E, H, C);
        k_colstats<<<HC, 256, 0, stream>>>(bufXR, bias, muB, varB, N, HC);
        k_bn_elu<<<g1((long)N * HC), 256, 0, stream>>>(
            bufXR, bias, muB, varB, gn, bn, Hout, (long)N, HC);
    };

    // layer 1: 15 (pad 32) -> 4x32 concat (HC=128)
    gat_layer(x,    15,  32,  c1Wl, c1bl, c1Wr, c1br, c1We, c1att, c1b, g1p, b1p, 4, 32,  bufH);
    // layer 2: 128 -> 4x64 concat (HC=256)
    gat_layer(bufH, 128, 128, c2Wl, c2bl, c2Wr, c2br, c2We, c2att, c2b, g2p, b2p, 4, 64,  bufH);
    // layer 3: 256 -> 1x256, concat=False (mean over 1 head == identity)
    gat_layer(bufH, 256, 256, c3Wl, c3bl, c3Wr, c3br, c3We, c3att, c3b, g3p, b3p, 1, 256, bufH);

    // graph mean-pool
    k_fill_f32<<<g1((long)G * 256 + G), 256, 0, stream>>>(gemb, 0.0f, (long)G * 256 + G);
    k_pool<<<g1((long)N * 256), 256, 0, stream>>>(bufH, batch, gemb, cnt, N, 256);
    k_pool_div<<<g1((long)G * 256), 256, 0, stream>>>(gemb, cnt, G, 256);

    float* outF      = (float*)d_out;
    float* out_value = outF;                       // [64]
    float* out_pol   = outF + G;                   // [64,4096]
    float* out_tac   = outF + G + (long)G * 4096;  // [64]
    float* out_str   = out_tac + G;                // [64]

    // heads: gemb -> bf16 once; Wv1 and Wp GEMMs share it (K=256)
    k_cvt_pad_bf16<<<g1((long)G * 256), 256, 0, stream>>>(gemb, Abf, G, 256, 256);
    pack_gemm(Abf, G, 256, 256, Wv1, bv1, hid, 128, 1);          // relu
    pack_gemm(Abf, G, 256, 256, Wp,  bp,  out_pol, 4096, 0);     // policy
    k_dot<<<g1((long)G * 32), 256, 0, stream>>>(hid, 128, Wv2, bv2, out_value, G, 1);
    k_dot<<<g1((long)G * 32), 256, 0, stream>>>(gemb, 256, Wt, bt, out_tac, G, 0);
    k_dot<<<g1((long)G * 32), 256, 0, stream>>>(gemb, 256, Ws, bs, out_str, G, 0);
}
